// GraphNeuralNetwork_62895501082697
// MI455X (gfx1250) — compile-verified, matrix operand order checked by reading, not run on hardware
//
#include <hip/hip_runtime.h>
#include <hip/hip_bf16.h>

// ---------------------------------------------------------------------------
// Types for CDNA5 WMMA
// ---------------------------------------------------------------------------
typedef __attribute__((ext_vector_type(16))) __bf16        v16bf;
typedef __attribute__((ext_vector_type(8)))  float         v8f;
typedef __attribute__((ext_vector_type(8)))  unsigned int  v8u;
typedef __attribute__((ext_vector_type(2)))  float         f32x2;
typedef __attribute__((ext_vector_type(2)))  __bf16        bf16x2;

#define D_DIM 256
#define M_DIM 64
#define R_DIM 12

// pack two f32 -> packed bf16 pair; vector convert lowers to v_cvt_pk_bf16_f32
__device__ __forceinline__ unsigned int pack_bf16(float a, float b) {
    f32x2 f = { a, b };
    bf16x2 h = __builtin_convertvector(f, bf16x2);
    return __builtin_bit_cast(unsigned int, h);
}

// ---------------------------------------------------------------------------
// WMMA GEMM:  C[rows x Dout] = epi( A[rows x K] @ B[K x Dout] + bias )
//   - 256 threads = 8 waves arranged 4 (rows) x 2 (cols)
//   - block tile 128 x BN_T, k-step 32; wave tile 32 x (BN_T/2)
//   - A/B fp32 in memory -> packed bf16 in LDS (ds_store_b64)
//   - register-staged pipeline: next tile's global loads issued before WMMA
//   - epilogue: optional bias, tanh-GELU, residual add
// ---------------------------------------------------------------------------
template <int BN_T>
__global__ __launch_bounds__(256) void gnn_wmma_gemm(
    const float* __restrict__ A, const float* __restrict__ B,
    const float* __restrict__ bias, const float* __restrict__ residual,
    float* __restrict__ C, int Nrows, int K, int Dout, int doGelu)
{
    constexpr int BM_T  = 128;
    constexpr int WN    = BN_T / 2;     // wave column tile
    constexpr int NT    = WN / 16;      // B frags per wave
    constexpr int BITER = BN_T / 32;    // B staging iterations

    // packed-bf16 tiles: [row][k-pair] ; row = 32 bf16 = 16 uints = 64 B
    __shared__ __align__(16) unsigned int As_u[BM_T][16];
    __shared__ __align__(16) unsigned int Bt_u[BN_T][16];   // [col][k-pair]

    const int tid   = threadIdx.x;
    const int wave  = tid >> 5;
    const int waveR = wave & 3;         // 0..3 -> 32-row group
    const int waveC = wave >> 2;        // 0..1 -> WN-col group
    const int lane  = tid & 31;
    const int half  = lane >> 4;
    const int l16   = lane & 15;

    const int rowBase = blockIdx.y * BM_T;
    const int colBase = blockIdx.x * BN_T;

    v8f acc[2][NT] = {};

    // staging coordinates (A: 1024 float4 over 256 threads; B: (col,kquad))
    int aRow[4], aCol[4];
#pragma unroll
    for (int it = 0; it < 4; ++it) {
        int lin  = tid + it * 256;        // 0..1023
        aRow[it] = lin >> 3;              // 0..127
        aCol[it] = (lin & 7) << 2;        // 0,4,..,28
    }
    int bCol[BITER], bK[BITER];
#pragma unroll
    for (int it = 0; it < BITER; ++it) {
        int lin  = tid + it * 256;        // 0..BN_T*8-1
        bCol[it] = lin & (BN_T - 1);
        bK[it]   = (lin >> (BN_T == 128 ? 7 : 6)) << 2;   // k-quad * 4
    }

    float4 aReg[4];
    float  bReg[BITER][4];

    auto loadTiles = [&](int k0) {
#pragma unroll
        for (int it = 0; it < 4; ++it) {
            int grow = rowBase + aRow[it];
            if (grow < Nrows)
                aReg[it] = *(const float4*)(A + (long long)grow * K + k0 + aCol[it]);
            else
                aReg[it] = make_float4(0.f, 0.f, 0.f, 0.f);
        }
#pragma unroll
        for (int it = 0; it < BITER; ++it) {
#pragma unroll
            for (int c = 0; c < 4; ++c)
                bReg[it][c] = B[(long long)(k0 + bK[it] + c) * Dout + colBase + bCol[it]];
        }
    };

    auto storeTiles = [&]() {
#pragma unroll
        for (int it = 0; it < 4; ++it) {
            uint2 p = make_uint2(pack_bf16(aReg[it].x, aReg[it].y),
                                 pack_bf16(aReg[it].z, aReg[it].w));
            *(uint2*)&As_u[aRow[it]][aCol[it] >> 1] = p;     // ds_store_b64
        }
#pragma unroll
        for (int it = 0; it < BITER; ++it) {
            uint2 p = make_uint2(pack_bf16(bReg[it][0], bReg[it][1]),
                                 pack_bf16(bReg[it][2], bReg[it][3]));
            *(uint2*)&Bt_u[bCol[it]][bK[it] >> 1] = p;       // ds_store_b64
        }
    };

    const int kTiles = K / 32;
    loadTiles(0);

    for (int kt = 0; kt < kTiles; ++kt) {
        __syncthreads();                 // LDS free (previous tile consumed)
        storeTiles();
        __syncthreads();

        if (kt + 1 < kTiles) loadTiles((kt + 1) * 32);   // overlap with WMMA

        // A fragments (16x32 bf16): lanes 0-15 K{0..7,16..23}, lanes 16-31 K{8..15,24..31}
        v16bf afrag[2];
#pragma unroll
        for (int at = 0; at < 2; ++at) {
            const int arow = waveR * 32 + at * 16 + l16;
            uint4 lo = *(const uint4*)&As_u[arow][half * 4];
            uint4 hi = *(const uint4*)&As_u[arow][8 + half * 4];
            v8u raw = { lo.x, lo.y, lo.z, lo.w, hi.x, hi.y, hi.z, hi.w };
            afrag[at] = __builtin_bit_cast(v16bf, raw);
        }
#pragma unroll
        for (int nt = 0; nt < NT; ++nt) {
            const int bcol = waveC * WN + nt * 16 + l16;
            uint4 lo = *(const uint4*)&Bt_u[bcol][half * 4];
            uint4 hi = *(const uint4*)&Bt_u[bcol][8 + half * 4];
            v8u raw = { lo.x, lo.y, lo.z, lo.w, hi.x, hi.y, hi.z, hi.w };
            v16bf bfrag = __builtin_bit_cast(v16bf, raw);
#pragma unroll
            for (int at = 0; at < 2; ++at)
                acc[at][nt] = __builtin_amdgcn_wmma_f32_16x16x32_bf16(
                    false, afrag[at], false, bfrag, (short)0, acc[at][nt], false, false);
        }
    }

    // epilogue; C layout: VGPR r, lanes 0-15 -> M=r, lanes 16-31 -> M=r+8
#pragma unroll
    for (int nt = 0; nt < NT; ++nt) {
        const int col  = colBase + waveC * WN + nt * 16 + l16;
        const float bv = bias ? bias[col] : 0.f;
#pragma unroll
        for (int at = 0; at < 2; ++at) {
#pragma unroll
            for (int r = 0; r < 8; ++r) {
                const int row = rowBase + waveR * 32 + at * 16 + r + half * 8;
                if (row < Nrows) {
                    float v = acc[at][nt][r] + bv;
                    if (doGelu) {
                        float u = 0.7978845608028654f * (v + 0.044715f * v * v * v);
                        v = 0.5f * v * (1.0f + tanhf(u));
                    }
                    if (residual) v += residual[(long long)row * Dout + col];
                    C[(long long)row * Dout + col] = v;
                }
            }
        }
    }
}

// ---------------------------------------------------------------------------
// Node embedding init: x0[n][d] = emb_table[z[n]] (H=5) @ emb_W + emb_b
// ---------------------------------------------------------------------------
__global__ void gnn_embed(const int* __restrict__ z, const float* __restrict__ table,
                          const float* __restrict__ W, const float* __restrict__ b,
                          float* __restrict__ out, int Nn)
{
    int gid = blockIdx.x * blockDim.x + threadIdx.x;
    int n = gid >> 8, d = gid & 255;
    if (n >= Nn) return;
    int t = z[n];
    float acc = b[d];
#pragma unroll
    for (int h = 0; h < 5; ++h)
        acc += table[t * 5 + h] * W[h * D_DIM + d];
    out[(long long)n * D_DIM + d] = acc;
}

// ---------------------------------------------------------------------------
// Bessel RBF with p=5 polynomial envelope: rbf[e][r] = env(x) * sin(freq_r * x)
// ---------------------------------------------------------------------------
__global__ void gnn_rbf(const float* __restrict__ pos, const int* __restrict__ ei,
                        const float* __restrict__ freqs, float* __restrict__ rbf, int E_)
{
    int e = blockIdx.x * blockDim.x + threadIdx.x;
    if (e >= E_) return;
    int i = ei[e], j = ei[E_ + e];
    float dx = pos[i * 3 + 0] - pos[j * 3 + 0];
    float dy = pos[i * 3 + 1] - pos[j * 3 + 1];
    float dz = pos[i * 3 + 2] - pos[j * 3 + 2];
    float dist = sqrtf(dx * dx + dy * dy + dz * dz + 1e-12f);
    float x = dist * (1.0f / 5.0f);
    float x2 = x * x;
    float x4 = x2 * x2;
    // a=-21 (x^4), b=35 (x^5), c=-15 (x^6)
    float env = 1.0f / x + x4 * (-21.0f + x * (35.0f + x * (-15.0f)));
#pragma unroll
    for (int r = 0; r < R_DIM; ++r)
        rbf[(long long)e * R_DIM + r] = env * sinf(freqs[r] * x);
}

__global__ void gnn_zero(float* __restrict__ p, int n)
{
    int gid = blockIdx.x * blockDim.x + threadIdx.x;
    if (gid < n) p[gid] = 0.f;
}

// ---------------------------------------------------------------------------
// Edge phase: m[e] = v[j[e]] * (rbf[e] @ Wrbf); agg[i[e]] += m[e]
// 16 threads per edge, 4 message channels each; scatter via fp32 atomics (L2).
// ---------------------------------------------------------------------------
__global__ void gnn_edge_msg(const float* __restrict__ rbf, const float* __restrict__ Wrbf,
                             const float* __restrict__ v, const int* __restrict__ ei,
                             float* __restrict__ agg, int E_)
{
    __shared__ float wr[R_DIM * M_DIM];   // 3 KB
    for (int t = threadIdx.x; t < R_DIM * M_DIM; t += blockDim.x) wr[t] = Wrbf[t];
    __syncthreads();

    int gid = blockIdx.x * blockDim.x + threadIdx.x;
    if (gid >= E_ * 16) return;
    int e = gid >> 4;
    int m = (gid & 15) << 2;            // 4 channels of M=64
    int i = ei[e], j = ei[E_ + e];

    // rbf row (12 floats) as 3x float4
    float rb[R_DIM];
    {
        const float4* rp = (const float4*)(rbf + (long long)e * R_DIM);
        float4 r0 = rp[0], r1 = rp[1], r2 = rp[2];
        rb[0] = r0.x; rb[1] = r0.y; rb[2]  = r0.z; rb[3]  = r0.w;
        rb[4] = r1.x; rb[5] = r1.y; rb[6]  = r1.z; rb[7]  = r1.w;
        rb[8] = r2.x; rb[9] = r2.y; rb[10] = r2.z; rb[11] = r2.w;
    }

    float g0 = 0.f, g1 = 0.f, g2 = 0.f, g3 = 0.f;
#pragma unroll
    for (int r = 0; r < R_DIM; ++r) {
        const float* w = &wr[r * M_DIM + m];
        g0 += rb[r] * w[0];
        g1 += rb[r] * w[1];
        g2 += rb[r] * w[2];
        g3 += rb[r] * w[3];
    }
    float4 vj = *(const float4*)(v + (long long)j * M_DIM + m);
    float* a = agg + (long long)i * M_DIM + m;
    atomicAdd(a + 0, vj.x * g0);
    atomicAdd(a + 1, vj.y * g1);
    atomicAdd(a + 2, vj.z * g2);
    atomicAdd(a + 3, vj.w * g3);
}

// ---------------------------------------------------------------------------
// Orchestration
// ---------------------------------------------------------------------------
extern "C" void kernel_launch(void* const* d_in, const int* in_sizes, int n_in,
                              void* d_out, int out_size, void* d_ws, size_t ws_size,
                              hipStream_t stream)
{
    const int*   z         = (const int*)  d_in[0];
    const float* pos       = (const float*)d_in[1];
    const int*   ei        = (const int*)  d_in[4];
    const float* emb_table = (const float*)d_in[5];
    const float* emb_W     = (const float*)d_in[6];
    const float* emb_b     = (const float*)d_in[7];
    const float* freqs     = (const float*)d_in[8];
    const float* fc0_W     = (const float*)d_in[9];
    const float* fc0_b     = (const float*)d_in[10];
    const float* conv_Wv   = (const float*)d_in[11];
    const float* conv_Wrbf = (const float*)d_in[12];
    const float* conv_Wout = (const float*)d_in[13];
    const float* fc_W      = (const float*)d_in[14];
    const float* fc_b      = (const float*)d_in[15];

    const int N  = in_sizes[0];        // z is (N,1)
    const int E_ = in_sizes[4] / 2;    // edge_index is (2,E)

    float* x   = (float*)d_out;                       // node state [N,256]
    float* h1  = (float*)d_ws;                        // [N,256]
    float* h2  = h1  + (size_t)N * D_DIM;             // [N,256]
    float* vb  = h2  + (size_t)N * D_DIM;             // [N,64]
    float* agg = vb  + (size_t)N * M_DIM;             // [N,64]
    float* rbf = agg + (size_t)N * M_DIM;             // [E,12]

    const dim3 blk(256);
    const dim3 gD(D_DIM / 128, (N + 127) / 128);      // Dout=256 -> <128> variant
    const dim3 gM(1, (N + 127) / 128);                // Dout=64  -> <64>  variant

    // init embedding + fc0 block (2x GELU GEMM), x lands in d_out
    gnn_embed<<<(N * D_DIM + 255) / 256, 256, 0, stream>>>(z, emb_table, emb_W, emb_b, h1, N);
    gnn_wmma_gemm<128><<<gD, blk, 0, stream>>>(h1, fc0_W,                 fc0_b,         nullptr, h2, N, D_DIM, D_DIM, 1);
    gnn_wmma_gemm<128><<<gD, blk, 0, stream>>>(h2, fc0_W + D_DIM * D_DIM, fc0_b + D_DIM, nullptr, x,  N, D_DIM, D_DIM, 1);

    // radial basis once
    gnn_rbf<<<(E_ + 255) / 256, 256, 0, stream>>>(pos, ei, freqs, rbf, E_);

    for (int n = 0; n < 4; ++n) {
        // v = x @ Wv[n]   [N,256]x[256,64]
        gnn_wmma_gemm<64><<<gM, blk, 0, stream>>>(x, conv_Wv + (size_t)n * D_DIM * M_DIM,
                                                  nullptr, nullptr, vb, N, D_DIM, M_DIM, 0);
        // agg = segment_sum(v[j] * (rbf @ Wrbf[n]), i)
        gnn_zero<<<(N * M_DIM + 255) / 256, 256, 0, stream>>>(agg, N * M_DIM);
        gnn_edge_msg<<<(E_ * 16 + 255) / 256, 256, 0, stream>>>(
            rbf, conv_Wrbf + (size_t)n * R_DIM * M_DIM, vb, ei, agg, E_);
        // x = x + agg @ Wout[n]   [N,64]x[64,256] with residual
        gnn_wmma_gemm<128><<<gD, blk, 0, stream>>>(agg, conv_Wout + (size_t)n * M_DIM * D_DIM,
                                                   nullptr, x, x, N, M_DIM, D_DIM, 0);
        // x = x + gelu(gelu(x@W0+b0)@W1+b1)
        gnn_wmma_gemm<128><<<gD, blk, 0, stream>>>(x,  fc_W + (size_t)(n * 2 + 0) * D_DIM * D_DIM,
                                                   fc_b + (n * 2 + 0) * D_DIM, nullptr, h1, N, D_DIM, D_DIM, 1);
        gnn_wmma_gemm<128><<<gD, blk, 0, stream>>>(h1, fc_W + (size_t)(n * 2 + 1) * D_DIM * D_DIM,
                                                   fc_b + (n * 2 + 1) * D_DIM, x,       x,  N, D_DIM, D_DIM, 1);
    }
}
